// GMAN_78400333021797
// MI455X (gfx1250) — matrix operand order repaired and checked by reading, not compiled, not run on hardware
//
#include <hip/hip_runtime.h>
#include <cstdint>
#include <cstddef>

// ---------------------------------------------------------------------------
// GMAN forward for MI455X (gfx1250, wave32). All dense layers run through
// v_wmma_f32_16x16x32_f16 (f16 operands, f32 accumulate). Problem is fully
// L2-resident (all tensors <= 8MB vs 192MB L2), so we optimize for matrix-op
// throughput and contiguous b128 fragment loads, not HBM streaming.
// ---------------------------------------------------------------------------

typedef _Float16 half_t;
typedef __attribute__((ext_vector_type(16))) _Float16 v16h;
typedef __attribute__((ext_vector_type(8)))  _Float16 v8h;
typedef __attribute__((ext_vector_type(8)))  float    v8f;
typedef __attribute__((ext_vector_type(4)))  int      v4i;

#define NB     8
#define TSTEPS 12
#define NNODE  325
#define NTOK   (NB*TSTEPS*NNODE)   // 31200 tokens
#define NKEY   352                 // 325 keys padded to 22*16

// CDNA5 async global->LDS path (ASYNCcnt-tracked), guarded so the build can
// never break if the toolchain lacks the builtins.
#if defined(__has_builtin)
#if __has_builtin(__builtin_amdgcn_global_load_async_to_lds_b128) && \
    __has_builtin(__builtin_amdgcn_s_wait_asynccnt)
#define HAVE_ASYNC_LDS 1
#endif
#endif

#ifdef HAVE_ASYNC_LDS
typedef __attribute__((address_space(1))) v4i* glb_v4ip;
typedef __attribute__((address_space(3))) v4i* lds_v4ip;
#endif

__device__ __forceinline__ v8f wmma16(v16h a, v16h b, v8f c) {
  // (neg_a, A, neg_b, B, c_mod, C, reuse_a, reuse_b)
  return __builtin_amdgcn_wmma_f32_16x16x32_f16(false, a, false, b, (short)0, c,
                                                false, false);
}

// --------------------------- weight packing --------------------------------
// W is (K x 64) row-major f32; pack to Wt (64 x K) row-major f16 so the WMMA
// B-fragment per lane is 16 contiguous halfs (two b128 loads).
__global__ void pack_wt(const float* __restrict__ W, half_t* __restrict__ Wt,
                        int K, int N) {
  int i = blockIdx.x * 256 + threadIdx.x;
  if (i >= K * N) return;
  int n = i % N, k = i / N;
  Wt[(size_t)n * K + k] = (half_t)W[(size_t)k * N + n];
}

// ----------------------------- generic GEMM --------------------------------
// C[M x 64] = act(A[M x KT] @ W[KT x 64] + bias) (+ addsrc). Templated on KT
// and A dtype so the K-loop fully unrolls into a straight WMMA stream.
template <int KT, int AF32>
__global__ __launch_bounds__(128) void gemm64(
    const void* __restrict__ Aptr, int lda, int M,
    const half_t* __restrict__ Wt,
    const float* __restrict__ bias, const float* __restrict__ addsrc,
    half_t* __restrict__ y16, float* __restrict__ yf32, int act) {
  const int lane = threadIdx.x & 31;
  const int wv   = threadIdx.x >> 5;
  const int m0   = blockIdx.x * 64 + wv * 16;
  if (m0 >= M) return;
  const int mr = lane & 15;    // row within tile (A) / column (B/C)
  const int hh = lane >> 4;    // lane half selects K sub-range
  int mA = m0 + mr; if (mA >= M) mA = M - 1;   // clamp; bad rows not stored

  v8f z8 = {0.f,0.f,0.f,0.f,0.f,0.f,0.f,0.f};
  v8f acc[4] = {z8, z8, z8, z8};

#pragma unroll
  for (int k0 = 0; k0 < KT; k0 += 32) {
    // A fragment (16x32): lane half hh carries K = k0+16*hh..+15
    v16h a;
    if (!AF32) {
      const half_t* Ar = (const half_t*)Aptr + (size_t)mA * lda;
      v8h p0 = *(const v8h*)(Ar + k0 + 16 * hh);
      v8h p1 = *(const v8h*)(Ar + k0 + 16 * hh + 8);
#pragma unroll
      for (int i = 0; i < 8; ++i) { a[i] = p0[i]; a[8 + i] = p1[i]; }
    } else {
      const float* Ar = (const float*)Aptr + (size_t)mA * lda;
      v8f f0 = *(const v8f*)(Ar + k0 + 16 * hh);
      v8f f1 = *(const v8f*)(Ar + k0 + 16 * hh + 8);
#pragma unroll
      for (int i = 0; i < 8; ++i) { a[i] = (half_t)f0[i]; a[8 + i] = (half_t)f1[i]; }
    }
#pragma unroll
    for (int t = 0; t < 4; ++t) {
      // B fragment (32x16): lane n = t*16+mr, 16 contiguous K values
      const half_t* wp = Wt + (size_t)(t * 16 + mr) * KT + k0 + 16 * hh;
      v16h b = *(const v16h*)wp;
      acc[t] = wmma16(a, b, acc[t]);
    }
  }
  // C layout: element (g + 8*hh, mr) per VGPR g
#pragma unroll
  for (int t = 0; t < 4; ++t) {
    const int n = t * 16 + mr;
    const float bn = bias ? bias[n] : 0.f;
#pragma unroll
    for (int g = 0; g < 8; ++g) {
      const int m = m0 + g + 8 * hh;
      if (m < M) {
        float v = acc[t][g] + bn;
        if (act) v = v > 0.f ? v : 0.f;
        size_t o = (size_t)m * 64 + n;
        if (addsrc) v += addsrc[o];
        if (yf32) yf32[o] = v;
        if (y16)  y16[o]  = (half_t)v;
      }
    }
  }
}

// ------------------------- spatial attention core --------------------------
// One wave per (b,t,head). K/V head slices staged in LDS (K rows via async
// global->LDS b128 when available); logits and O=P@V via WMMA (d=8 zero-
// padded into K=32); two-pass softmax over 325 keys.
__global__ __launch_bounds__(32) void spatial_attn(
    const half_t* __restrict__ Q, const half_t* __restrict__ Kk,
    const half_t* __restrict__ V, half_t* __restrict__ O) {
  const int h  = blockIdx.x & 7;
  const int bt = blockIdx.x >> 3;
  const size_t base = (size_t)bt * NNODE * 64 + h * 8;

  __shared__ __align__(16) half_t Klds[NKEY * 8];      // [key][dd]
  __shared__ __align__(32) half_t Vt[16 * NKEY];       // [dd][key], rows 8..15 = 0
  __shared__ float  S[16 * NKEY];                      // logits / probs

  const int tid = threadIdx.x;
#ifdef HAVE_ASYNC_LDS
  // Async-stage K rows: 16B global row -> 16B LDS row, tracked by ASYNCcnt.
  for (int key = tid; key < NNODE; key += 32) {
    __builtin_amdgcn_global_load_async_to_lds_b128(
        (glb_v4ip)(uintptr_t)(Kk + base + (size_t)key * 64),
        (lds_v4ip)(uintptr_t)(&Klds[key * 8]), 0, 0);
  }
  for (int key = tid; key < NKEY; key += 32) {
    if (key >= NNODE) {
      v8h z = {};
      *(v8h*)&Klds[key * 8] = z;
    }
    v8h vv = {};
    if (key < NNODE) vv = *(const v8h*)(V + base + (size_t)key * 64);
#pragma unroll
    for (int dd = 0; dd < 8; ++dd) {
      Vt[dd * NKEY + key]       = vv[dd];
      Vt[(dd + 8) * NKEY + key] = (half_t)0.f;
    }
  }
  __builtin_amdgcn_s_wait_asynccnt(0);
#else
  for (int key = tid; key < NKEY; key += 32) {
    v8h kv = {}, vv = {};
    if (key < NNODE) {
      kv = *(const v8h*)(Kk + base + (size_t)key * 64);
      vv = *(const v8h*)(V  + base + (size_t)key * 64);
    }
    *(v8h*)&Klds[key * 8] = kv;
#pragma unroll
    for (int dd = 0; dd < 8; ++dd) {
      Vt[dd * NKEY + key]       = vv[dd];
      Vt[(dd + 8) * NKEY + key] = (half_t)0.f;
    }
  }
#endif
  __syncthreads();

  const int mr = tid & 15, hh = tid >> 4;
  const float scale = 0.35355339059327373f;  // 1/sqrt(8)
  v8f z8 = {0.f,0.f,0.f,0.f,0.f,0.f,0.f,0.f};

  for (int qt = 0; qt < 22; ++qt) {
    int qrow = qt * 16 + mr;
    int qc = qrow < NNODE ? qrow : NNODE - 1;
    // A frag = Q tile (16 x 32): only lanes 0-15, elements 0..7 carry d=0..7
    v16h a = {};
    if (hh == 0) {
      v8h qp = *(const v8h*)(Q + base + (size_t)qc * 64);
#pragma unroll
      for (int i = 0; i < 8; ++i) a[i] = qp[i];
    }
    // logits vs all key tiles
    for (int kt = 0; kt < 22; ++kt) {
      v16h b = {};
      if (hh == 0) {
        v8h kp = *(const v8h*)&Klds[(kt * 16 + mr) * 8];
#pragma unroll
        for (int i = 0; i < 8; ++i) b[i] = kp[i];
      }
      v8f c = wmma16(a, b, z8);
#pragma unroll
      for (int g = 0; g < 8; ++g)
        S[(g + 8 * hh) * NKEY + kt * 16 + mr] = c[g] * scale;
    }
    __syncthreads();
    // softmax: lane pair (mr, mr+16) handles row mr, strided keys
    float mx = -1e30f;
    for (int key = hh; key < NNODE; key += 2) {
      float v = S[mr * NKEY + key];
      mx = v > mx ? v : mx;
    }
    mx = fmaxf(mx, __shfl_xor(mx, 16));
    float sum = 0.f;
    for (int key = hh; key < NNODE; key += 2) {
      float e = __expf(S[mr * NKEY + key] - mx);
      S[mr * NKEY + key] = e; sum += e;
    }
    sum += __shfl_xor(sum, 16);
    float inv = 1.f / sum;
    for (int key = hh; key < NNODE; key += 2) S[mr * NKEY + key] *= inv;
    for (int key = NNODE + hh; key < NKEY; key += 2) S[mr * NKEY + key] = 0.f;
    __syncthreads();
    // O = P @ V  (K = 352 keys, 11 WMMA steps)
    v8f oc = z8;
#pragma unroll
    for (int k0 = 0; k0 < NKEY; k0 += 32) {
      v16h pa;
#pragma unroll
      for (int i = 0; i < 8; ++i) {
        pa[i]     = (half_t)S[mr * NKEY + k0 + 16 * hh + i];
        pa[8 + i] = (half_t)S[mr * NKEY + k0 + 16 * hh + 8 + i];
      }
      v16h vb = *(const v16h*)&Vt[mr * NKEY + k0 + 16 * hh];  // column dd = mr
      oc = wmma16(pa, vb, oc);
    }
    if (mr < 8) {
#pragma unroll
      for (int g = 0; g < 8; ++g) {
        int m = qt * 16 + g + 8 * hh;
        if (m < NNODE) O[base + (size_t)m * 64 + mr] = (half_t)oc[g];
      }
    }
    __syncthreads();
  }
}

// ---------------- temporal (causal) & transform attention (T=12) ----------
template <int CAUSAL>
__global__ __launch_bounds__(96) void small_attn(
    const half_t* __restrict__ Q, const half_t* __restrict__ Kk,
    const half_t* __restrict__ V, half_t* __restrict__ O) {
  int bn = blockIdx.x; int b = bn / NNODE, n = bn % NNODE;
  int tid = threadIdx.x; int h = tid / TSTEPS, t = tid % TSTEPS;
  const size_t st = (size_t)NNODE * 64;
  size_t base = (size_t)b * TSTEPS * st + (size_t)n * 64 + h * 8;
  v8h qv = *(const v8h*)(Q + base + (size_t)t * st);
  float q[8];
#pragma unroll
  for (int d2 = 0; d2 < 8; ++d2) q[d2] = (float)qv[d2];
  const int send = CAUSAL ? (t + 1) : TSTEPS;
  float lg[TSTEPS]; float mx = -1e30f;
  for (int s = 0; s < send; ++s) {
    v8h kv = *(const v8h*)(Kk + base + (size_t)s * st);
    float acc = 0.f;
#pragma unroll
    for (int d2 = 0; d2 < 8; ++d2) acc += q[d2] * (float)kv[d2];
    acc *= 0.35355339059327373f; lg[s] = acc; mx = fmaxf(mx, acc);
  }
  float sum = 0.f;
  for (int s = 0; s < send; ++s) { lg[s] = __expf(lg[s] - mx); sum += lg[s]; }
  float inv = 1.f / sum;
  float o[8] = {0,0,0,0,0,0,0,0};
  for (int s = 0; s < send; ++s) {
    float p = lg[s] * inv;
    v8h vv = *(const v8h*)(V + base + (size_t)s * st);
#pragma unroll
    for (int d2 = 0; d2 < 8; ++d2) o[d2] += p * (float)vv[d2];
  }
#pragma unroll
  for (int d2 = 0; d2 < 8; ++d2) O[base + (size_t)t * st + d2] = (half_t)o[d2];
}

// ----------------------------- small helpers -------------------------------
__global__ void te_embed(const float* __restrict__ TE, const float* __restrict__ W1,
                         const float* __restrict__ W2, float* __restrict__ TEe) {
  int row = blockIdx.x;       // 192 = B*(P+Q)
  int c = threadIdx.x;        // 64
  __shared__ float sh[64];
  float acc = 0.f;
  for (int i = 0; i < 32; ++i) acc += TE[row * 32 + i] * W1[i * 64 + c];
  sh[c] = acc > 0.f ? acc : 0.f;
  __syncthreads();
  float out = 0.f;
  for (int k = 0; k < 64; ++k) out += sh[k] * W2[k * 64 + c];
  TEe[row * 64 + c] = out;
}

__global__ void input_embed(const float* __restrict__ X, const float* __restrict__ W1,
                            const float* __restrict__ b1, half_t* __restrict__ A16) {
  int i = blockIdx.x * 256 + threadIdx.x;
  if (i >= NTOK * 64) return;
  int c = i & 63; int token = i >> 6;
  float v = X[token] * W1[c] + b1[c];
  A16[i] = (half_t)(v > 0.f ? v : 0.f);
}

__global__ void build_xc(const float* __restrict__ embX, const float* __restrict__ TEe,
                         const float* __restrict__ se, int teoff,
                         half_t* __restrict__ A16) {
  int i = blockIdx.x * 256 + threadIdx.x;
  if (i >= NTOK * 64) return;
  int c = i & 63; int token = i >> 6;
  int n = token % NNODE; int bt = token / NNODE;
  int t = bt % TSTEPS; int b = bt / TSTEPS;
  A16[(size_t)token * 128 + c] = (half_t)embX[i];
  A16[(size_t)token * 128 + 64 + c] =
      (half_t)(TEe[(b * 24 + teoff + t) * 64 + c] + se[n * 64 + c]);
}

__global__ void build_ste(const float* __restrict__ TEe, const float* __restrict__ se,
                          int teoff, half_t* __restrict__ A16) {
  int i = blockIdx.x * 256 + threadIdx.x;
  if (i >= NTOK * 64) return;
  int c = i & 63; int token = i >> 6;
  int n = token % NNODE; int bt = token / NNODE;
  int t = bt % TSTEPS; int b = bt / TSTEPS;
  A16[i] = (half_t)(TEe[(b * 24 + teoff + t) * 64 + c] + se[n * 64 + c]);
}

__global__ void conv_f16(const float* __restrict__ src, half_t* __restrict__ dst, int ntot) {
  int i = blockIdx.x * 256 + threadIdx.x;
  if (i < ntot) dst[i] = (half_t)src[i];
}

__global__ void fuse_gate(const float* __restrict__ G, const float* __restrict__ HS,
                          const float* __restrict__ HT, half_t* __restrict__ H16, int ntot) {
  int i = blockIdx.x * 256 + threadIdx.x;
  if (i >= ntot) return;
  float z = 1.f / (1.f + __expf(-G[i]));
  H16[i] = (half_t)(z * HS[i] + (1.f - z) * HT[i]);
}

__global__ void final_out(const half_t* __restrict__ Hid, const float* __restrict__ W2,
                          const float* __restrict__ b2, float* __restrict__ out) {
  int token = blockIdx.x * 256 + threadIdx.x;
  if (token >= NTOK) return;
  float acc = 0.f;
#pragma unroll
  for (int c = 0; c < 8; ++c) {
    v8h hv = *(const v8h*)(Hid + (size_t)token * 64 + c * 8);
#pragma unroll
    for (int i = 0; i < 8; ++i) acc += (float)hv[i] * W2[c * 8 + i];
  }
  out[token] = acc + b2[0];
}

// ------------------------------- host side ---------------------------------
extern "C" void kernel_launch(void* const* d_in, const int* in_sizes, int n_in,
                              void* d_out, int out_size, void* d_ws, size_t ws_size,
                              hipStream_t stream) {
  (void)in_sizes; (void)n_in; (void)out_size; (void)ws_size;
  auto inf = [&](int i) -> const float* { return (const float*)d_in[i]; };

  char* wsbase = (char*)d_ws;
  size_t cur = 0;
  auto alloc = [&](size_t bytes) -> void* {
    cur = (cur + 255) & ~(size_t)255;
    void* p = wsbase + cur; cur += bytes; return p;
  };
  auto packW = [&](int idx, int K) -> half_t* {
    half_t* dst = (half_t*)alloc((size_t)K * 64 * sizeof(half_t));
    int tot = K * 64;
    pack_wt<<<(tot + 255) / 256, 256, 0, stream>>>(inf(idx), dst, K, 64);
    return dst;
  };
  struct PA { half_t *WQ,*WK,*WV,*WX1,*WX2; const float *bQ,*bK,*bV,*bX1,*bX2; int K; };
  auto packAttn = [&](int b0, int K) -> PA {
    PA p; p.K = K;
    p.WQ = packW(b0 + 0, K); p.bQ = inf(b0 + 1);
    p.WK = packW(b0 + 2, K); p.bK = inf(b0 + 3);
    p.WV = packW(b0 + 4, K); p.bV = inf(b0 + 5);
    p.WX1 = packW(b0 + 6, 64); p.bX1 = inf(b0 + 7);
    p.WX2 = packW(b0 + 8, 64); p.bX2 = inf(b0 + 9);
    return p;
  };
  struct PG { half_t *WS,*WT,*WH1,*WH2; const float *bT,*bH1,*bH2; };
  auto packGF = [&](int g0) -> PG {
    PG g;
    g.WS = packW(g0 + 0, 64); g.WT = packW(g0 + 1, 64); g.bT = inf(g0 + 2);
    g.WH1 = packW(g0 + 3, 64); g.bH1 = inf(g0 + 4);
    g.WH2 = packW(g0 + 5, 64); g.bH2 = inf(g0 + 6);
    return g;
  };

  // Input index map (insertion order of setup_inputs / _make_params):
  // 0:X 1:TE 2:te_W1 3:te_W2 4:se 5:in_W1 6:in_b1 7:in_W2 8:in_b2
  // enc0:9..35 enc1:36..62 trans:63..72 dec0:73..99 dec1:100..126
  // 127:out_W1 128:out_b1 129:out_W2 130:out_b2
  half_t* inW2 = packW(7, 64);
  PA encSA[2], encTA[2], decSA[2], decTA[2]; PG encGF[2], decGF[2];
  encSA[0] = packAttn(9, 128);  encTA[0] = packAttn(19, 128);  encGF[0] = packGF(29);
  encSA[1] = packAttn(36, 128); encTA[1] = packAttn(46, 128);  encGF[1] = packGF(56);
  PA tr = packAttn(63, 64);
  decSA[0] = packAttn(73, 128); decTA[0] = packAttn(83, 128);  decGF[0] = packGF(93);
  decSA[1] = packAttn(100, 128); decTA[1] = packAttn(110, 128); decGF[1] = packGF(120);
  half_t* outW1 = packW(127, 64);

  // Activations (~52 MB total, heavy reuse).
  float*  TEe  = (float*)alloc(192 * 64 * sizeof(float));
  float*  embX = (float*)alloc((size_t)NTOK * 64 * sizeof(float));
  float*  HS   = (float*)alloc((size_t)NTOK * 64 * sizeof(float));
  float*  HT   = (float*)alloc((size_t)NTOK * 64 * sizeof(float));
  half_t* A16  = (half_t*)alloc((size_t)NTOK * 128 * sizeof(half_t));
  half_t* B16  = (half_t*)alloc((size_t)NTOK * 64 * sizeof(half_t));
  half_t* Q16  = (half_t*)alloc((size_t)NTOK * 64 * sizeof(half_t));
  half_t* K16  = (half_t*)alloc((size_t)NTOK * 64 * sizeof(half_t));
  half_t* V16  = (half_t*)alloc((size_t)NTOK * 64 * sizeof(half_t));
  half_t* O16  = (half_t*)alloc((size_t)NTOK * 64 * sizeof(half_t));
  float*  T1   = (float*)A16;   // gate pre-activation reuses Xc bytes

  const float* X  = inf(0);
  const float* TE = inf(1);
  const float* se = inf(4);

  const int EW = (NTOK * 64 + 255) / 256;   // elementwise grid
  const dim3 GG((NTOK + 63) / 64), GB(128);
  // K=128 f16-A, K=64 f16-A, K=64 f32-A specializations
  auto gemm128 = [&](const half_t* A, half_t* W, const float* bi, const float* add,
                     half_t* y16, float* y32, int act) {
    gemm64<128,0><<<GG, GB, 0, stream>>>(A, 128, NTOK, W, bi, add, y16, y32, act);
  };
  auto gemm64h = [&](const half_t* A, half_t* W, const float* bi, const float* add,
                     half_t* y16, float* y32, int act) {
    gemm64<64,0><<<GG, GB, 0, stream>>>(A, 64, NTOK, W, bi, add, y16, y32, act);
  };
  auto gemm64f = [&](const float* A, half_t* W, const float* bi, const float* add,
                     half_t* y16, float* y32, int act) {
    gemm64<64,1><<<GG, GB, 0, stream>>>(A, 64, NTOK, W, bi, add, y16, y32, act);
  };

  // STE embeddings + input projection
  te_embed<<<192, 64, 0, stream>>>(TE, inf(2), inf(3), TEe);
  input_embed<<<EW, 256, 0, stream>>>(X, inf(5), inf(6), A16);
  gemm64h(A16, inW2, inf(8), nullptr, nullptr, embX, 0);

  auto runBlock = [&](PA& SA, PA& TA, PG& GF, int teoff) {
    build_xc<<<EW, 256, 0, stream>>>(embX, TEe, se, teoff, A16);
    // spatial attention branch
    gemm128(A16, SA.WQ, SA.bQ, nullptr, Q16, nullptr, 1);
    gemm128(A16, SA.WK, SA.bK, nullptr, K16, nullptr, 1);
    gemm128(A16, SA.WV, SA.bV, nullptr, V16, nullptr, 1);
    spatial_attn<<<NB * TSTEPS * 8, 32, 0, stream>>>(Q16, K16, V16, O16);
    gemm64h(O16, SA.WX1, SA.bX1, nullptr, B16, nullptr, 1);
    gemm64h(B16, SA.WX2, SA.bX2, nullptr, nullptr, HS, 0);
    // temporal attention branch
    gemm128(A16, TA.WQ, TA.bQ, nullptr, Q16, nullptr, 1);
    gemm128(A16, TA.WK, TA.bK, nullptr, K16, nullptr, 1);
    gemm128(A16, TA.WV, TA.bV, nullptr, V16, nullptr, 1);
    small_attn<1><<<NB * NNODE, 96, 0, stream>>>(Q16, K16, V16, O16);
    gemm64h(O16, TA.WX1, TA.bX1, nullptr, B16, nullptr, 1);
    gemm64h(B16, TA.WX2, TA.bX2, nullptr, nullptr, HT, 0);
    // gated fusion + residual (Xc in A16 no longer needed -> T1 aliases it)
    gemm64f(HS, GF.WS, nullptr, nullptr, nullptr, T1, 0);
    gemm64f(HT, GF.WT, GF.bT, T1, nullptr, T1, 0);
    fuse_gate<<<EW, 256, 0, stream>>>(T1, HS, HT, O16, NTOK * 64);
    gemm64h(O16, GF.WH1, GF.bH1, nullptr, B16, nullptr, 1);
    gemm64h(B16, GF.WH2, GF.bH2, embX, nullptr, embX, 0);
  };

  runBlock(encSA[0], encTA[0], encGF[0], 0);
  runBlock(encSA[1], encTA[1], encGF[1], 0);

  // transform attention (queries from STE_Q, keys from STE_P, values from embX)
  build_ste<<<EW, 256, 0, stream>>>(TEe, se, 12, A16);
  gemm64h(A16, tr.WQ, tr.bQ, nullptr, Q16, nullptr, 1);
  build_ste<<<EW, 256, 0, stream>>>(TEe, se, 0, A16);
  gemm64h(A16, tr.WK, tr.bK, nullptr, K16, nullptr, 1);
  conv_f16<<<EW, 256, 0, stream>>>(embX, A16, NTOK * 64);
  gemm64h(A16, tr.WV, tr.bV, nullptr, V16, nullptr, 1);
  small_attn<0><<<NB * NNODE, 96, 0, stream>>>(Q16, K16, V16, O16);
  gemm64h(O16, tr.WX1, tr.bX1, nullptr, B16, nullptr, 1);
  gemm64h(B16, tr.WX2, tr.bX2, nullptr, nullptr, embX, 0);

  runBlock(decSA[0], decTA[0], decGF[0], 12);
  runBlock(decSA[1], decTA[1], decGF[1], 12);

  // output head
  conv_f16<<<EW, 256, 0, stream>>>(embX, A16, NTOK * 64);
  gemm64h(A16, outW1, inf(128), nullptr, B16, nullptr, 1);
  final_out<<<(NTOK + 255) / 256, 256, 0, stream>>>(B16, inf(129), inf(130),
                                                    (float*)d_out);
}
